// DebertaBarePinyin_39926015984259
// MI455X (gfx1250) — matrix operand order repaired and checked by reading, not compiled, not run on hardware
//
#include <hip/hip_runtime.h>
#include <hip/hip_bf16.h>

// ---------------------------------------------------------------------------
// DeBERTa-pinyin encoder forward for MI455X (gfx1250, wave32, WMMA).
// All GEMMs use v_wmma_f32_16x16x32_bf16 with fp32 accumulation.
// 64x64 C tile / 128-thread block; each wave: 2x2 WMMA subtiles.
// Weights are pre-transposed to n-major bf16 once per layer so every GEMM
// B-operand stages with straight b128 copies (K contiguous).
// ---------------------------------------------------------------------------

typedef __attribute__((ext_vector_type(16))) __bf16 v16bf;
typedef __attribute__((ext_vector_type(8)))  __bf16 v8bf;
typedef __attribute__((ext_vector_type(8)))  float  v8f;

#define SQ   512
#define DM   768
#define NH   12
#define NB   4
#define NP   512     // rel-embedding rows (2*BUCKETS)
#define DHD  64

// ---------------------------------------------------------------------------
// Relative-position bucket indices (DeBERTa-v2 log bucketing), device-side.
// ---------------------------------------------------------------------------
__global__ void relpos_kernel(int* __restrict__ c2pi, int* __restrict__ p2ci) {
  int idx = blockIdx.x * blockDim.x + threadIdx.x;
  if (idx >= SQ * SQ) return;
  int q = idx >> 9, k = idx & 511;
  int rel = q - k;
  const int mid = 128;                       // BUCKETS/2
  int sgn = (rel > 0) - (rel < 0);
  int abs_pos = (rel < mid && rel > -mid) ? (mid - 1) : (rel < 0 ? -rel : rel);
  float log_pos = ceilf(logf((float)abs_pos / (float)mid) /
                        logf(511.0f / (float)mid) * (float)(mid - 1)) + (float)mid;
  int bucket = (abs_pos <= mid) ? rel : (int)(log_pos * (float)sgn);
  int c = bucket + 256; c = c < 0 ? 0 : (c > 511 ? 511 : c);
  int p = -bucket + 256; p = p < 0 ? 0 : (p > 511 ? 511 : p);
  c2pi[idx] = c;
  p2ci[idx] = p;
}

// ---------------------------------------------------------------------------
// fp32 -> bf16 convert (for rel_emb).
// ---------------------------------------------------------------------------
__global__ void f32_to_bf16_kernel(const float* __restrict__ x, __bf16* __restrict__ y, int n) {
  int i = blockIdx.x * blockDim.x + threadIdx.x;
  if (i < n) y[i] = (__bf16)x[i];
}

// ---------------------------------------------------------------------------
// Weight transpose + convert: Wt[n][k] = (bf16)W[k][n].
// 32x32 tiles via LDS, 256 threads; coalesced on both sides.
// ---------------------------------------------------------------------------
__global__ __launch_bounds__(256)
void transpose_w_kernel(const float* __restrict__ W, __bf16* __restrict__ Wt,
                        int K, int N) {
  __shared__ float tile[32][33];
  int nb = blockIdx.x * 32, kb = blockIdx.y * 32;
  int tx = threadIdx.x & 31, ty = threadIdx.x >> 5;   // ty = 0..7
#pragma unroll
  for (int i = 0; i < 32; i += 8)
    tile[ty + i][tx] = W[(long)(kb + ty + i) * N + nb + tx];
  __syncthreads();
#pragma unroll
  for (int i = 0; i < 32; i += 8)
    Wt[(long)(nb + ty + i) * K + kb + tx] = (__bf16)tile[tx][ty + i];
}

// ---------------------------------------------------------------------------
// Embedding: gather tok+pinyin+seg, LayerNorm, mask multiply. One block/token.
// ---------------------------------------------------------------------------
__global__ __launch_bounds__(256)
void embed_kernel(const int* __restrict__ ids, const int* __restrict__ segs,
                  const int* __restrict__ pins, const float* __restrict__ amask,
                  const float* __restrict__ tokE, const float* __restrict__ pinE,
                  const float* __restrict__ segE, const float* __restrict__ lnw,
                  const float* __restrict__ lnb, float* __restrict__ h,
                  __bf16* __restrict__ hbf) {
  int t = blockIdx.x, tid = threadIdx.x;
  int id = ids[t], sg = segs[t], pn = pins[t];
  float m = amask[t];
  float vals[3]; float s = 0.f, s2 = 0.f;
#pragma unroll
  for (int j = 0; j < 3; ++j) {
    int c = tid + 256 * j;
    float v = tokE[(long)id * DM + c] + pinE[(long)pn * DM + c] + segE[(long)sg * DM + c];
    vals[j] = v; s += v; s2 += v * v;
  }
  __shared__ float r1[256], r2[256];
  r1[tid] = s; r2[tid] = s2; __syncthreads();
  for (int off = 128; off > 0; off >>= 1) {
    if (tid < off) { r1[tid] += r1[tid + off]; r2[tid] += r2[tid + off]; }
    __syncthreads();
  }
  float mu  = r1[0] * (1.f / DM);
  float var = r2[0] * (1.f / DM) - mu * mu;
  float inv = rsqrtf(var + 1e-12f);
#pragma unroll
  for (int j = 0; j < 3; ++j) {
    int c = tid + 256 * j;
    float y = ((vals[j] - mu) * inv * lnw[c] + lnb[c]) * m;
    h[(long)t * DM + c] = y;
    hbf[(long)t * DM + c] = (__bf16)y;
  }
}

// ---------------------------------------------------------------------------
// Residual add + LayerNorm (fp32 + bf16 outputs). One block/token.
// ---------------------------------------------------------------------------
__global__ __launch_bounds__(256)
void add_ln_kernel(const float* __restrict__ x, const float* __restrict__ resid,
                   const float* __restrict__ lnw, const float* __restrict__ lnb,
                   float* __restrict__ out, __bf16* __restrict__ outbf) {
  int t = blockIdx.x, tid = threadIdx.x;
  float vals[3]; float s = 0.f, s2 = 0.f;
#pragma unroll
  for (int j = 0; j < 3; ++j) {
    int c = tid + 256 * j;
    float v = x[(long)t * DM + c] + resid[(long)t * DM + c];
    vals[j] = v; s += v; s2 += v * v;
  }
  __shared__ float r1[256], r2[256];
  r1[tid] = s; r2[tid] = s2; __syncthreads();
  for (int off = 128; off > 0; off >>= 1) {
    if (tid < off) { r1[tid] += r1[tid + off]; r2[tid] += r2[tid + off]; }
    __syncthreads();
  }
  float mu  = r1[0] * (1.f / DM);
  float var = r2[0] * (1.f / DM) - mu * mu;
  float inv = rsqrtf(var + 1e-12f);
#pragma unroll
  for (int j = 0; j < 3; ++j) {
    int c = tid + 256 * j;
    float y = (vals[j] - mu) * inv * lnw[c] + lnb[c];
    out[(long)t * DM + c] = y;
    outbf[(long)t * DM + c] = (__bf16)y;
  }
}

// ---------------------------------------------------------------------------
// Disentangled-attention softmax: scores + gather(c2p) + gather(p2c^T) + mask,
// row softmax over keys, bf16 probs. One block (512 threads) per (b,h,q) row.
// ---------------------------------------------------------------------------
__global__ __launch_bounds__(512)
void attn_softmax_kernel(const float* __restrict__ scores, const float* __restrict__ c2p,
                         const float* __restrict__ p2c, const int* __restrict__ c2pi,
                         const int* __restrict__ p2ci, const float* __restrict__ amask,
                         __bf16* __restrict__ probs) {
  int r = blockIdx.x;                 // (b*NH + h)*SQ + q
  int q = r & (SQ - 1);
  int bh = r >> 9;                    // b*NH + h
  int b = bh / NH;
  int k = threadIdx.x;
  long srow = (long)r * SQ;
  long bhoff = (long)bh * SQ * SQ;
  float s = scores[srow + k];
  s += c2p[bhoff + (long)q * NP + c2pi[q * SQ + k]];
  s += p2c[bhoff + (long)k * NP + p2ci[k * SQ + q]];
  s += (1.f - amask[b * SQ + k]) * -1e9f;
  __shared__ float red[512];
  red[k] = s; __syncthreads();
  for (int off = 256; off > 0; off >>= 1) {
    if (k < off) red[k] = fmaxf(red[k], red[k + off]);
    __syncthreads();
  }
  float mx = red[0];
  __syncthreads();
  float e = __expf(s - mx);
  red[k] = e; __syncthreads();
  for (int off = 256; off > 0; off >>= 1) {
    if (k < off) red[k] += red[k + off];
    __syncthreads();
  }
  float inv = 1.f / red[0];
  probs[srow + k] = (__bf16)(e * inv);
}

// ---------------------------------------------------------------------------
// WMMA GEMM: C[m,n] = act( scale * sum_k A[m,k]*B[k,n] + bias[n] )
//  - 128 threads (4 waves), 64x64 C tile; wave quadrant = 2x2 WMMA subtiles.
//  - LDS A tile [m][k]; LDS B tile n-major [n][k] so every fragment read is
//    two ds_load_b128 per lane (contiguous K runs per CDNA5 WMMA layout).
//  - BT: B source is n-major bf16 (transposed weights / K^T / pos^T):
//    straight b128 staging.  !BT: bf16 k-major (v matrix): transpose-scatter.
//  - Batched over grid.z = b*Hdim + h with independent strides.
// ---------------------------------------------------------------------------
template <bool BT>
__global__ __launch_bounds__(128)
void gemm_wmma_kernel(const __bf16* __restrict__ A, const __bf16* __restrict__ B,
                      float* __restrict__ Cf, __bf16* __restrict__ Cb,
                      const float* __restrict__ bias,
                      int M, int N, int K, int lda, int ldb, int ldc,
                      long sAb, long sAh, long sBb, long sBh, long sCb, long sCh,
                      int Hdim, float scale, int act_gelu) {
  int zb = blockIdx.z / Hdim, zh = blockIdx.z % Hdim;
  A += zb * sAb + zh * sAh;
  B += zb * sBb + zh * sBh;
  long coff = zb * sCb + zh * sCh;

  int n0 = blockIdx.x * 64, m0 = blockIdx.y * 64;
  int tid  = threadIdx.x;
  int lane = tid & 31, wave = tid >> 5;
  int wm = wave >> 1, wn = wave & 1;     // wave quadrant of the 64x64 tile
  int lm = lane & 15, kh = lane >> 4;

  __shared__ __align__(16) __bf16 As[64][40];  // [m][k], padded rows
  __shared__ __align__(16) __bf16 Bs[64][40];  // [n][k], padded rows

  v8f acc[2][2] = {};

  int r2 = tid >> 1, h2 = tid & 1;       // vector staging coords (64 rows x 2)
  int kk = tid & 31, ng = tid >> 5;      // scatter staging coords (!BT only)

  for (int k0 = 0; k0 < K; k0 += 32) {
    // ---- stage A tile: 64 m x 32 k, two b128 per thread ----
    {
      const __bf16* src = A + (long)(m0 + r2) * lda + k0 + 16 * h2;
      *(uint4*)&As[r2][16 * h2]     = *(const uint4*)src;
      *(uint4*)&As[r2][16 * h2 + 8] = *(const uint4*)(src + 8);
      if (k0 + 32 < K) __builtin_prefetch(src + 32, 0, 1);
    }
    // ---- stage B tile into n-major LDS ----
    if constexpr (BT) {
      // source already n-major, contiguous in k: direct vector copy
      const __bf16* src = B + (long)(n0 + r2) * ldb + k0 + 16 * h2;
      *(uint4*)&Bs[r2][16 * h2]     = *(const uint4*)src;
      *(uint4*)&Bs[r2][16 * h2 + 8] = *(const uint4*)(src + 8);
      if (k0 + 32 < K) __builtin_prefetch(src + 32, 0, 1);
    } else {
      // bf16 k-major (v matrix): transpose-scatter
      const __bf16* src = B + (long)(k0 + kk) * ldb + n0 + 16 * ng;
      uint4 d0 = *(const uint4*)src;
      uint4 d1 = *(const uint4*)(src + 8);
      const __bf16* ds0 = (const __bf16*)&d0;
      const __bf16* ds1 = (const __bf16*)&d1;
#pragma unroll
      for (int i = 0; i < 8; ++i) Bs[16 * ng + i][kk] = ds0[i];
#pragma unroll
      for (int i = 0; i < 8; ++i) Bs[16 * ng + 8 + i][kk] = ds1[i];
      if (k0 + 32 < K) __builtin_prefetch(src + (long)32 * ldb, 0, 1);
    }
    __syncthreads();

    // ---- fragments: all reads are contiguous b128 pairs ----
    v16bf fa[2], fb[2];
#pragma unroll
    for (int mi = 0; mi < 2; ++mi) {
      const __bf16* p = &As[32 * wm + 16 * mi + lm][kh * 8];
      ((v8bf*)&fa[mi])[0] = *(const v8bf*)p;         // K = kh*8 + 0..7
      ((v8bf*)&fa[mi])[1] = *(const v8bf*)(p + 16);  // K = 16 + kh*8 + 0..7
    }
#pragma unroll
    for (int ni = 0; ni < 2; ++ni) {
      const __bf16* p = &Bs[32 * wn + 16 * ni + lm][kh * 8];
      ((v8bf*)&fb[ni])[0] = *(const v8bf*)p;         // K = kh*8 + 0..7
      ((v8bf*)&fb[ni])[1] = *(const v8bf*)(p + 16);  // K = 16 + kh*8 + 0..7
    }
#pragma unroll
    for (int mi = 0; mi < 2; ++mi)
#pragma unroll
      for (int ni = 0; ni < 2; ++ni)
        acc[mi][ni] = __builtin_amdgcn_wmma_f32_16x16x32_bf16(
            false, fa[mi], false, fb[ni], (short)0, acc[mi][ni], false, false);
    __syncthreads();
  }

  // ---- epilogue: scale, bias, optional exact GELU, fp32/bf16 stores ----
#pragma unroll
  for (int mi = 0; mi < 2; ++mi)
#pragma unroll
    for (int ni = 0; ni < 2; ++ni)
#pragma unroll
      for (int j = 0; j < 8; ++j) {
        int m = m0 + 32 * wm + 16 * mi + 8 * kh + j;
        int n = n0 + 32 * wn + 16 * ni + lm;
        float v = acc[mi][ni][j] * scale;
        if (bias) v += bias[n];
        if (act_gelu) v = 0.5f * v * (1.f + erff(v * 0.70710678118654752f));
        long ci = coff + (long)m * ldc + n;
        if (Cf) Cf[ci] = v;
        if (Cb) Cb[ci] = (__bf16)v;
      }
}

// ---------------------------------------------------------------------------
// Host-side orchestration.
// ---------------------------------------------------------------------------
extern "C" void kernel_launch(void* const* d_in, const int* in_sizes, int n_in,
                              void* d_out, int out_size, void* d_ws, size_t ws_size,
                              hipStream_t stream) {
  (void)in_sizes; (void)n_in; (void)out_size; (void)ws_size;

  const int*   input_ids   = (const int*)d_in[0];
  const int*   segment_ids = (const int*)d_in[1];
  const int*   pinyin_ids  = (const int*)d_in[2];
  const float* amask       = (const float*)d_in[3];
  const float* tokE        = (const float*)d_in[4];
  const float* pinE        = (const float*)d_in[5];
  const float* segE        = (const float*)d_in[6];
  const float* elnw        = (const float*)d_in[7];
  const float* elnb        = (const float*)d_in[8];
  const float* relE        = (const float*)d_in[9];
  const float* Wq = (const float*)d_in[10]; const float* bq = (const float*)d_in[11];
  const float* Wk = (const float*)d_in[12]; const float* bk = (const float*)d_in[13];
  const float* Wv = (const float*)d_in[14]; const float* bv = (const float*)d_in[15];
  const float* Wo = (const float*)d_in[16]; const float* bo = (const float*)d_in[17];
  const float* ln1w = (const float*)d_in[18]; const float* ln1b = (const float*)d_in[19];
  const float* W1 = (const float*)d_in[20]; const float* b1 = (const float*)d_in[21];
  const float* W2 = (const float*)d_in[22]; const float* b2 = (const float*)d_in[23];
  const float* ln2w = (const float*)d_in[24]; const float* ln2b = (const float*)d_in[25];

  const int L = 12, F = 3072, M = NB * SQ;  // M = 2048 tokens

  // ---- carve workspace ----
  size_t off = 0;
  auto carve = [&](size_t bytes) -> void* {
    void* p = (char*)d_ws + off;
    off += (bytes + 255) & ~(size_t)255;
    return p;
  };
  float*  h      = (float*)carve((size_t)M * DM * 4);
  float*  tmp    = (float*)carve((size_t)M * DM * 4);
  __bf16* hbf    = (__bf16*)carve((size_t)M * DM * 2);
  __bf16* qbf    = (__bf16*)carve((size_t)M * DM * 2);
  __bf16* kbf    = (__bf16*)carve((size_t)M * DM * 2);
  __bf16* vbf    = (__bf16*)carve((size_t)M * DM * 2);
  __bf16* ctxbf  = (__bf16*)carve((size_t)M * DM * 2);
  __bf16* relbf  = (__bf16*)carve((size_t)NP * DM * 2);
  __bf16* poskbf = (__bf16*)carve((size_t)NP * DM * 2);
  __bf16* posqbf = (__bf16*)carve((size_t)NP * DM * 2);
  __bf16* ffbf   = (__bf16*)carve((size_t)M * F * 2);
  __bf16* wqT    = (__bf16*)carve((size_t)DM * DM * 2);  // [n][k] bf16 weights
  __bf16* wkT    = (__bf16*)carve((size_t)DM * DM * 2);
  __bf16* wvT    = (__bf16*)carve((size_t)DM * DM * 2);
  __bf16* woT    = (__bf16*)carve((size_t)DM * DM * 2);
  __bf16* w1T    = (__bf16*)carve((size_t)DM * F * 2);   // [F][DM]
  __bf16* w2T    = (__bf16*)carve((size_t)F * DM * 2);   // [DM][F]
  float*  scores = (float*)carve((size_t)NB * NH * SQ * SQ * 4);
  float*  c2p    = (float*)carve((size_t)NB * NH * SQ * NP * 4);
  float*  p2c    = (float*)carve((size_t)NB * NH * SQ * NP * 4);
  __bf16* probs  = (__bf16*)carve((size_t)NB * NH * SQ * SQ * 2);
  int*    c2pi   = (int*)carve((size_t)SQ * SQ * 4);
  int*    p2ci   = (int*)carve((size_t)SQ * SQ * 4);

  const float scaleAttn = 1.0f / sqrtf((float)(DHD * 3));
  const long sBS = (long)SQ * DM;       // per-batch stride in token tensors
  const long sHS = (long)SQ * SQ;       // per-head stride in score tensors

  // ---- one-time prep ----
  relpos_kernel<<<(SQ * SQ + 255) / 256, 256, 0, stream>>>(c2pi, p2ci);
  f32_to_bf16_kernel<<<(NP * DM + 255) / 256, 256, 0, stream>>>(relE, relbf, NP * DM);
  embed_kernel<<<M, 256, 0, stream>>>(input_ids, segment_ids, pinyin_ids, amask,
                                      tokE, pinE, segE, elnw, elnb, h, hbf);

  for (int i = 0; i < L; ++i) {
    const float* Wq_i = Wq + (size_t)i * DM * DM; const float* bq_i = bq + (size_t)i * DM;
    const float* Wk_i = Wk + (size_t)i * DM * DM; const float* bk_i = bk + (size_t)i * DM;
    const float* Wv_i = Wv + (size_t)i * DM * DM; const float* bv_i = bv + (size_t)i * DM;
    const float* Wo_i = Wo + (size_t)i * DM * DM; const float* bo_i = bo + (size_t)i * DM;
    const float* W1_i = W1 + (size_t)i * DM * F;  const float* b1_i = b1 + (size_t)i * F;
    const float* W2_i = W2 + (size_t)i * F * DM;  const float* b2_i = b2 + (size_t)i * DM;

    // --- pre-transpose this layer's weights to n-major bf16 ---
    dim3 gT(DM / 32, DM / 32);
    transpose_w_kernel<<<gT, 256, 0, stream>>>(Wq_i, wqT, DM, DM);
    transpose_w_kernel<<<gT, 256, 0, stream>>>(Wk_i, wkT, DM, DM);
    transpose_w_kernel<<<gT, 256, 0, stream>>>(Wv_i, wvT, DM, DM);
    transpose_w_kernel<<<gT, 256, 0, stream>>>(Wo_i, woT, DM, DM);
    transpose_w_kernel<<<dim3(F / 32, DM / 32), 256, 0, stream>>>(W1_i, w1T, DM, F);
    transpose_w_kernel<<<dim3(DM / 32, F / 32), 256, 0, stream>>>(W2_i, w2T, F, DM);

    // --- Q/K/V projections: [2048,768] = hbf @ W + b  -> bf16 ---
    dim3 gProj(DM / 64, M / 64, 1);
    gemm_wmma_kernel<true><<<gProj, 128, 0, stream>>>(
        hbf, wqT, nullptr, qbf, bq_i, M, DM, DM, DM, DM, DM,
        0, 0, 0, 0, 0, 0, 1, 1.0f, 0);
    gemm_wmma_kernel<true><<<gProj, 128, 0, stream>>>(
        hbf, wkT, nullptr, kbf, bk_i, M, DM, DM, DM, DM, DM,
        0, 0, 0, 0, 0, 0, 1, 1.0f, 0);
    gemm_wmma_kernel<true><<<gProj, 128, 0, stream>>>(
        hbf, wvT, nullptr, vbf, bv_i, M, DM, DM, DM, DM, DM,
        0, 0, 0, 0, 0, 0, 1, 1.0f, 0);

    // --- relative-position projections: [512,768] = relbf @ Wk/Wq + b ---
    dim3 gPos(DM / 64, NP / 64, 1);
    gemm_wmma_kernel<true><<<gPos, 128, 0, stream>>>(
        relbf, wkT, nullptr, poskbf, bk_i, NP, DM, DM, DM, DM, DM,
        0, 0, 0, 0, 0, 0, 1, 1.0f, 0);
    gemm_wmma_kernel<true><<<gPos, 128, 0, stream>>>(
        relbf, wqT, nullptr, posqbf, bq_i, NP, DM, DM, DM, DM, DM,
        0, 0, 0, 0, 0, 0, 1, 1.0f, 0);

    // --- scores[b,h,q,k] = scale * q . k   (batched over b*NH) ---
    dim3 gScore(SQ / 64, SQ / 64, NB * NH);
    gemm_wmma_kernel<true><<<gScore, 128, 0, stream>>>(
        qbf, kbf, scores, nullptr, nullptr, SQ, SQ, DHD, DM, DM, SQ,
        sBS, DHD, sBS, DHD, (long)NH * sHS, sHS, NH, scaleAttn, 0);

    // --- c2p[b,h,q,p] = scale * q . pos_k ---
    gemm_wmma_kernel<true><<<gScore, 128, 0, stream>>>(
        qbf, poskbf, c2p, nullptr, nullptr, SQ, NP, DHD, DM, DM, NP,
        sBS, DHD, 0, DHD, (long)NH * sHS, sHS, NH, scaleAttn, 0);

    // --- p2c[b,h,k,p] = scale * k . pos_q ---
    gemm_wmma_kernel<true><<<gScore, 128, 0, stream>>>(
        kbf, posqbf, p2c, nullptr, nullptr, SQ, NP, DHD, DM, DM, NP,
        sBS, DHD, 0, DHD, (long)NH * sHS, sHS, NH, scaleAttn, 0);

    // --- softmax with disentangled gathers + mask -> bf16 probs ---
    attn_softmax_kernel<<<NB * NH * SQ, 512, 0, stream>>>(
        scores, c2p, p2c, c2pi, p2ci, amask, probs);

    // --- ctx[b,q,h,d] = probs @ v  (batched over b*NH) -> bf16 ---
    dim3 gCtx(DHD / 64, SQ / 64, NB * NH);
    gemm_wmma_kernel<false><<<gCtx, 128, 0, stream>>>(
        probs, vbf, nullptr, ctxbf, nullptr, SQ, DHD, SQ, SQ, DM, DM,
        (long)NH * sHS, sHS, sBS, DHD, sBS, DHD, NH, 1.0f, 0);

    // --- output projection, then residual + LN1 ---
    gemm_wmma_kernel<true><<<gProj, 128, 0, stream>>>(
        ctxbf, woT, tmp, nullptr, bo_i, M, DM, DM, DM, DM, DM,
        0, 0, 0, 0, 0, 0, 1, 1.0f, 0);
    add_ln_kernel<<<M, 256, 0, stream>>>(tmp, h, ln1w + (size_t)i * DM,
                                         ln1b + (size_t)i * DM, h, hbf);

    // --- FFN: gelu(h @ W1 + b1) @ W2 + b2, residual + LN2 ---
    dim3 gFf1(F / 64, M / 64, 1);
    gemm_wmma_kernel<true><<<gFf1, 128, 0, stream>>>(
        hbf, w1T, nullptr, ffbf, b1_i, M, F, DM, DM, DM, F,
        0, 0, 0, 0, 0, 0, 1, 1.0f, 1);
    dim3 gFf2(DM / 64, M / 64, 1);
    gemm_wmma_kernel<true><<<gFf2, 128, 0, stream>>>(
        ffbf, w2T, tmp, nullptr, b2_i, M, DM, F, F, F, DM,
        0, 0, 0, 0, 0, 0, 1, 1.0f, 0);

    float* hout = (i == L - 1) ? (float*)d_out : h;
    add_ln_kernel<<<M, 256, 0, stream>>>(tmp, h, ln2w + (size_t)i * DM,
                                         ln2b + (size_t)i * DM, hout, hbf);
  }
}
